// BiMambaBlock_76493367542168
// MI455X (gfx1250) — matrix operand order repaired and checked by reading, not compiled, not run on hardware
//
#include <hip/hip_runtime.h>
#include <hip/hip_bf16.h>

// ---------------------------------------------------------------------------
// BiMamba block for MI455X (gfx1250, wave32).
//  - GEMMs: v_wmma_f32_16x16x32_bf16, 128x128 block tile, 32x64 per wave.
//  - LDS staging via CDNA5 async copies (global_load_async_to_lds_b128,
//    ASYNCcnt) with double buffering and one barrier per K-step.
// ---------------------------------------------------------------------------

typedef __attribute__((ext_vector_type(16))) __bf16 v16bf;
typedef __attribute__((ext_vector_type(8)))  float  v8f;

#define D_MODEL 768
#define D_STATE 16
#define D_CONV  4
#define D_INNER 1536
#define DT_RANK 48
#define B_SZ    2
#define SEQ     1024
#define M_ROWS  (B_SZ * SEQ)             // 2048
#define N_IN    (2 * D_INNER)            // 3072
#define N_XP    (DT_RANK + 2 * D_STATE)  // 80

static __device__ __forceinline__ __bf16 f2bf(float f) {
    union { float f; unsigned u; } v; v.f = f;
    unsigned r = (v.u + 0x7FFFu + ((v.u >> 16) & 1u)) >> 16;
    unsigned short s = (unsigned short)r;
    return __builtin_bit_cast(__bf16, s);
}

// ---------------------------------------------------------------------------
// f32 -> bf16 conversion
// ---------------------------------------------------------------------------
__global__ __launch_bounds__(256)
void cvt_f32_bf16_kernel(const float* __restrict__ src, __bf16* __restrict__ dst, int n) {
    int i = blockIdx.x * 256 + threadIdx.x;
    if (i < n) dst[i] = f2bf(src[i]);
}

// ---------------------------------------------------------------------------
// LayerNorm: writes bf16 normalized rows (forward order + time-flipped copy)
// ---------------------------------------------------------------------------
__global__ __launch_bounds__(256)
void layernorm_kernel(const float* __restrict__ x, const float* __restrict__ g,
                      const float* __restrict__ b, __bf16* __restrict__ xn,
                      __bf16* __restrict__ xnrev) {
    const int m = blockIdx.x;
    const int tid = threadIdx.x;
    __shared__ float red[256];

    float s = 0.f;
    for (int c = tid; c < D_MODEL; c += 256) s += x[(size_t)m * D_MODEL + c];
    red[tid] = s; __syncthreads();
    for (int st = 128; st > 0; st >>= 1) { if (tid < st) red[tid] += red[tid + st]; __syncthreads(); }
    const float mu = red[0] / (float)D_MODEL;
    __syncthreads();

    float v = 0.f;
    for (int c = tid; c < D_MODEL; c += 256) {
        float d = x[(size_t)m * D_MODEL + c] - mu; v += d * d;
    }
    red[tid] = v; __syncthreads();
    for (int st = 128; st > 0; st >>= 1) { if (tid < st) red[tid] += red[tid + st]; __syncthreads(); }
    const float inv = rsqrtf(red[0] / (float)D_MODEL + 1e-5f);

    const int bb = m / SEQ, t = m % SEQ;
    const int mr = bb * SEQ + (SEQ - 1 - t);
    for (int c = tid; c < D_MODEL; c += 256) {
        float val = (x[(size_t)m * D_MODEL + c] - mu) * inv * g[c] + b[c];
        __bf16 h = f2bf(val);
        xn[(size_t)m * D_MODEL + c] = h;
        xnrev[(size_t)mr * D_MODEL + c] = h;
    }
}

// ---------------------------------------------------------------------------
// WMMA GEMM:  C[M,N] = A[M,K] * Bw[N,K]^T   (both bf16, K-contiguous)
//   128x128 block tile, BK=32, 8 waves (4Mx2N), 32x64 per wave (2x4 frags).
//   Tiles staged with async global->LDS copies, double buffered (ASYNCcnt).
//   Epilogue: optional bias[n], residual, output-row time-flip, f32/bf16
//   stores with column offset col0 and row stride ldc.
// ---------------------------------------------------------------------------
#define TBM 128
#define TBN 128
#define TBK 32

__global__ __launch_bounds__(256)
void wmma_gemm_bf16_kernel(const __bf16* __restrict__ A, const __bf16* __restrict__ Bw,
                           float* Cf, __bf16* Cb,
                           int M, int N, int K, int ldc, int col0,
                           const float* bias, const float* resid, int flipL) {
    __shared__ __align__(16) __bf16 sA[2][TBM * TBK];
    __shared__ __align__(16) __bf16 sB[2][TBN * TBK];

    const int tid   = threadIdx.x;
    const int lane  = tid & 31;
    const int wave  = tid >> 5;
    const int wm    = wave & 3;   // wave M slot: 0..3  (32 rows each)
    const int wn    = wave >> 2;  // wave N slot: 0..1  (64 cols each)
    const int row16 = lane & 15;  // A: M row / B,C: N col
    const int kh    = lane >> 4;  // K-half (A/B) / +8 M rows (C)
    const int blockM = blockIdx.y * TBM;
    const int blockN = blockIdx.x * TBN;

    // per-thread staging chunks: 2x A (16B) + 2x B (16B) per K-step
    const int rA  = tid >> 1;              // rows 0..127, 2 chunks/row pair
    const int kcA0 = (tid & 1) * 16;       // 2 chunks of 8 bf16 -> cols 0..15 / 16..31

    auto stage = [&](int k0, int buf) {
#pragma unroll
        for (int j = 0; j < 2; ++j) {
            int kc = kcA0 + j * 8;
            unsigned      lds = (unsigned)(size_t)(void*)&sA[buf][rA * TBK + kc];
            unsigned long long ga =
                (unsigned long long)(size_t)(const void*)&A[(size_t)(blockM + rA) * K + k0 + kc];
            asm volatile("global_load_async_to_lds_b128 %0, %1, off"
                         :: "v"(lds), "v"(ga) : "memory");
        }
        int gn = blockN + rA;
        if (gn < N) {
#pragma unroll
            for (int j = 0; j < 2; ++j) {
                int kc = kcA0 + j * 8;
                unsigned      lds = (unsigned)(size_t)(void*)&sB[buf][rA * TBK + kc];
                unsigned long long ga =
                    (unsigned long long)(size_t)(const void*)&Bw[(size_t)gn * K + k0 + kc];
                asm volatile("global_load_async_to_lds_b128 %0, %1, off"
                             :: "v"(lds), "v"(ga) : "memory");
            }
        }
    };

    v8f acc[2][4] = {};

    int buf = 0;
    stage(0, 0);
    for (int k0 = 0; k0 < K; k0 += TBK) {
        asm volatile("s_wait_asynccnt 0" ::: "memory");  // own async copies landed
        __syncthreads();                                 // everyone's landed
        if (k0 + TBK < K) stage(k0 + TBK, buf ^ 1);      // prefetch next tile

        v16bf af[2], bfr[4];
#pragma unroll
        for (int mi = 0; mi < 2; ++mi) {
            const __bf16* ap = &sA[buf][(wm * 32 + mi * 16 + row16) * TBK];
#pragma unroll
            for (int i = 0; i < 16; ++i)
                af[mi][i] = ap[kh * 16 + i];           // K = kh*16 .. +15
        }
#pragma unroll
        for (int ni = 0; ni < 4; ++ni) {
            const __bf16* bp = &sB[buf][(wn * 64 + ni * 16 + row16) * TBK];
#pragma unroll
            for (int i = 0; i < 16; ++i)
                bfr[ni][i] = bp[kh * 16 + i];          // K = kh*16 .. +15
        }
#pragma unroll
        for (int mi = 0; mi < 2; ++mi)
#pragma unroll
            for (int ni = 0; ni < 4; ++ni)
                acc[mi][ni] = __builtin_amdgcn_wmma_f32_16x16x32_bf16(
                    false, af[mi], false, bfr[ni], (short)0, acc[mi][ni], false, false);
        buf ^= 1;
    }

    // --- epilogue ---
#pragma unroll
    for (int mi = 0; mi < 2; ++mi)
#pragma unroll
        for (int ni = 0; ni < 4; ++ni)
#pragma unroll
            for (int r = 0; r < 8; ++r) {
                int gm = blockM + wm * 32 + mi * 16 + kh * 8 + r;
                int gn = blockN + wn * 64 + ni * 16 + row16;
                if (gn >= N) continue;
                float v = acc[mi][ni][r];
                if (bias) v += bias[gn];
                int om = gm;
                if (flipL) {
                    int bb = gm / flipL, t = gm - bb * flipL;
                    om = bb * flipL + (flipL - 1 - t);
                }
                size_t oidx = (size_t)om * ldc + col0 + gn;
                if (resid) v += resid[oidx];
                if (Cf) Cf[oidx] = v;
                if (Cb) Cb[oidx] = f2bf(v);
            }
}

// ---------------------------------------------------------------------------
// Causal depthwise conv (width 4) + SiLU; reads xi half of xz (f32),
// writes xs in f32 (for scan) and bf16 (for xproj GEMM).
// ---------------------------------------------------------------------------
__global__ __launch_bounds__(256)
void conv_silu_kernel(const float* __restrict__ xz, const float* __restrict__ cw,
                      const float* __restrict__ cb,
                      float* __restrict__ xs_f, __bf16* __restrict__ xs_b) {
    int idx = blockIdx.x * 256 + threadIdx.x;            // over M_ROWS * D_INNER
    if (idx >= M_ROWS * D_INNER) return;
    int m = idx / D_INNER, d = idx - m * D_INNER;
    int b = m / SEQ, t = m - b * SEQ;
    float acc = cb[d];
#pragma unroll
    for (int j = 0; j < D_CONV; ++j) {
        int tt = t - (D_CONV - 1) + j;
        if (tt >= 0)
            acc += cw[d * D_CONV + j] * xz[(size_t)(b * SEQ + tt) * N_IN + d];
    }
    float s = acc / (1.f + expf(-acc));
    xs_f[idx] = s;
    xs_b[idx] = f2bf(s);
}

// ---------------------------------------------------------------------------
// delta = softplus(dt @ dt_w^T + dt_b); dt row (48) staged in LDS per m.
// grid = (D_INNER/256, M_ROWS)
// ---------------------------------------------------------------------------
__global__ __launch_bounds__(256)
void delta_kernel(const float* __restrict__ xdbl, const float* __restrict__ dt_w,
                  const float* __restrict__ dt_b, float* __restrict__ delta) {
    __shared__ float sdt[DT_RANK];
    const int m = blockIdx.y;
    const int d = blockIdx.x * 256 + threadIdx.x;
    if (threadIdx.x < DT_RANK) sdt[threadIdx.x] = xdbl[(size_t)m * N_XP + threadIdx.x];
    __syncthreads();
    float acc = dt_b[d];
#pragma unroll 8
    for (int r = 0; r < DT_RANK; ++r) acc += sdt[r] * dt_w[d * DT_RANK + r];
    float sp = (acc > 20.f) ? acc : log1pf(expf(acc));
    delta[(size_t)m * D_INNER + d] = sp;
}

// ---------------------------------------------------------------------------
// Selective scan: one thread per (batch, channel), 16-state recurrence,
// B/C vectors staged in LDS each timestep.  Applies +xs*D and SiLU(z) gate,
// writes bf16 for the out-projection GEMM.  grid = (D_INNER/256, B_SZ)
// ---------------------------------------------------------------------------
__global__ __launch_bounds__(256)
void scan_kernel(const float* __restrict__ xdbl, const float* __restrict__ delta,
                 const float* __restrict__ xs, const float* __restrict__ xz,
                 const float* __restrict__ A_log, const float* __restrict__ Dp,
                 __bf16* __restrict__ ygate) {
    const int b = blockIdx.y;
    const int d = blockIdx.x * 256 + threadIdx.x;
    __shared__ float sBC[2 * D_STATE];

    float a[D_STATE], h[D_STATE];
#pragma unroll
    for (int n = 0; n < D_STATE; ++n) {
        a[n] = -expf(A_log[d * D_STATE + n]);
        h[n] = 0.f;
    }
    const float Dv = Dp[d];

    for (int t = 0; t < SEQ; ++t) {
        const size_t m = (size_t)b * SEQ + t;
        if (threadIdx.x < 2 * D_STATE)
            sBC[threadIdx.x] = xdbl[m * N_XP + DT_RANK + threadIdx.x];
        __syncthreads();

        const float dlt = delta[m * D_INNER + d];
        const float xv  = xs[m * D_INNER + d];
        float y = 0.f;
#pragma unroll
        for (int n = 0; n < D_STATE; ++n) {
            float dA = expf(dlt * a[n]);
            h[n] = dA * h[n] + dlt * sBC[n] * xv;
            y += h[n] * sBC[D_STATE + n];
        }
        y += xv * Dv;
        const float zv = xz[m * N_IN + D_INNER + d];
        y *= zv / (1.f + expf(-zv));
        ygate[m * D_INNER + d] = f2bf(y);
        __syncthreads();
    }
}

// ---------------------------------------------------------------------------
// Host-side orchestration
// ---------------------------------------------------------------------------
static inline size_t take(size_t& off, size_t bytes) {
    size_t p = off;
    off = (off + bytes + 255) & ~(size_t)255;
    return p;
}

extern "C" void kernel_launch(void* const* d_in, const int* in_sizes, int n_in,
                              void* d_out, int out_size, void* d_ws, size_t ws_size,
                              hipStream_t stream) {
    (void)in_sizes; (void)n_in; (void)out_size; (void)ws_size;
    const float* x      = (const float*)d_in[0];
    const float* norm_g = (const float*)d_in[1];
    const float* norm_b = (const float*)d_in[2];
    const float* proj_w = (const float*)d_in[3];
    const float* proj_b = (const float*)d_in[4];
    // per-direction params: f at 5..13, b at 14..22
    // [in_w, conv_w, conv_b, xproj_w, dt_w, dt_b, A_log, D, out_w]

    char* ws = (char*)d_ws;
    size_t off = 0;
    __bf16* bf_in[2]  = { (__bf16*)(ws + take(off, (size_t)N_IN * D_MODEL * 2)),
                          (__bf16*)(ws + take(off, (size_t)N_IN * D_MODEL * 2)) };
    __bf16* bf_xp[2]  = { (__bf16*)(ws + take(off, (size_t)N_XP * D_INNER * 2)),
                          (__bf16*)(ws + take(off, (size_t)N_XP * D_INNER * 2)) };
    __bf16* bf_ow[2]  = { (__bf16*)(ws + take(off, (size_t)D_MODEL * D_INNER * 2)),
                          (__bf16*)(ws + take(off, (size_t)D_MODEL * D_INNER * 2)) };
    __bf16* bf_pw     =   (__bf16*)(ws + take(off, (size_t)D_MODEL * 2 * D_MODEL * 2));
    __bf16* xn_bf     =   (__bf16*)(ws + take(off, (size_t)M_ROWS * D_MODEL * 2));
    __bf16* xnr_bf    =   (__bf16*)(ws + take(off, (size_t)M_ROWS * D_MODEL * 2));
    float*  xz        =   (float*) (ws + take(off, (size_t)M_ROWS * N_IN * 4));
    float*  xs_f      =   (float*) (ws + take(off, (size_t)M_ROWS * D_INNER * 4));
    __bf16* xs_bf     =   (__bf16*)(ws + take(off, (size_t)M_ROWS * D_INNER * 2));
    float*  xdbl      =   (float*) (ws + take(off, (size_t)M_ROWS * N_XP * 4));
    float*  delta     =   (float*) (ws + take(off, (size_t)M_ROWS * D_INNER * 4));
    __bf16* yg_bf     =   (__bf16*)(ws + take(off, (size_t)M_ROWS * D_INNER * 2));
    __bf16* comb_bf   =   (__bf16*)(ws + take(off, (size_t)M_ROWS * 2 * D_MODEL * 2));

    auto cvt = [&](const float* s, __bf16* d, int n) {
        cvt_f32_bf16_kernel<<<(n + 255) / 256, 256, 0, stream>>>(s, d, n);
    };

    // --- weight conversions (f32 -> bf16) ---
    for (int dir = 0; dir < 2; ++dir) {
        const int base = 5 + dir * 9;
        cvt((const float*)d_in[base + 0], bf_in[dir], N_IN * D_MODEL);
        cvt((const float*)d_in[base + 3], bf_xp[dir], N_XP * D_INNER);
        cvt((const float*)d_in[base + 8], bf_ow[dir], D_MODEL * D_INNER);
    }
    cvt(proj_w, bf_pw, D_MODEL * 2 * D_MODEL);

    // --- LayerNorm (forward + flipped copies) ---
    layernorm_kernel<<<M_ROWS, 256, 0, stream>>>(x, norm_g, norm_b, xn_bf, xnr_bf);

    // --- per-direction pipeline ---
    for (int dir = 0; dir < 2; ++dir) {
        const int base = 5 + dir * 9;
        const float* conv_w = (const float*)d_in[base + 1];
        const float* conv_b = (const float*)d_in[base + 2];
        const float* dt_w   = (const float*)d_in[base + 4];
        const float* dt_b   = (const float*)d_in[base + 5];
        const float* A_log  = (const float*)d_in[base + 6];
        const float* Dp     = (const float*)d_in[base + 7];
        const __bf16* Ain   = dir ? xnr_bf : xn_bf;

        // in_proj: (2048,768) x (3072,768)^T -> xz (2048,3072) f32
        wmma_gemm_bf16_kernel<<<dim3(N_IN / TBN, M_ROWS / TBM), 256, 0, stream>>>(
            Ain, bf_in[dir], xz, nullptr, M_ROWS, N_IN, D_MODEL, N_IN, 0,
            nullptr, nullptr, 0);

        // causal conv + silu
        conv_silu_kernel<<<(M_ROWS * D_INNER + 255) / 256, 256, 0, stream>>>(
            xz, conv_w, conv_b, xs_f, xs_bf);

        // xproj: (2048,1536) x (80,1536)^T -> xdbl (2048,80) f32
        wmma_gemm_bf16_kernel<<<dim3((N_XP + TBN - 1) / TBN, M_ROWS / TBM), 256, 0, stream>>>(
            xs_bf, bf_xp[dir], xdbl, nullptr, M_ROWS, N_XP, D_INNER, N_XP, 0,
            nullptr, nullptr, 0);

        // delta = softplus(dt @ dt_w^T + dt_b)
        delta_kernel<<<dim3(D_INNER / 256, M_ROWS), 256, 0, stream>>>(
            xdbl, dt_w, dt_b, delta);

        // selective scan + D skip + SiLU(z) gate
        scan_kernel<<<dim3(D_INNER / 256, B_SZ), 256, 0, stream>>>(
            xdbl, delta, xs_f, xz, A_log, Dp, yg_bf);

        // out_proj: (2048,1536) x (768,1536)^T -> comb bf16 cols [dir*768, +768)
        // backward direction un-flips the time axis on store.
        wmma_gemm_bf16_kernel<<<dim3(D_MODEL / TBN, M_ROWS / TBM), 256, 0, stream>>>(
            yg_bf, bf_ow[dir], nullptr, comb_bf, M_ROWS, D_MODEL, D_INNER,
            2 * D_MODEL, dir * D_MODEL, nullptr, nullptr, dir ? SEQ : 0);
    }

    // --- final: comb (2048,1536) x proj_w (768,1536)^T + proj_b + x -> d_out ---
    wmma_gemm_bf16_kernel<<<dim3(D_MODEL / TBN, M_ROWS / TBM), 256, 0, stream>>>(
        comb_bf, bf_pw, (float*)d_out, nullptr, M_ROWS, D_MODEL, 2 * D_MODEL,
        D_MODEL, 0, proj_b, x, 0);
}
